// SNN_22797686408017
// MI455X (gfx1250) — matrix operand order repaired and checked by reading, not compiled, not run on hardware
//
#include <hip/hip_runtime.h>

// ---------------------------------------------------------------------------
// Persistent SNN (LIF) kernel for gfx1250 (MI455X).
//   x:[1000,512,128] f32 -> L1(250) -> L2(250) -> L3(2); LIF after each layer.
//   - Weights held in VGPRs as WMMA B-fragments for the whole T loop.
//   - Membrane potentials held in WMMA f32 accumulators (C operand carries
//     beta*m + bias - reset); spikes exchanged between layers via LDS (f16).
//   - x_t tile staged with gfx1250 async global->LDS copies (ASYNCcnt),
//     issued one step ahead and overlapped with layer-2/3 compute.
//   - A-fragments double-buffered in registers to avoid WMMA->VALU WAR
//     hazard nops (ISA 7.12.1).
// ---------------------------------------------------------------------------

#ifndef __has_builtin
#define __has_builtin(x) 0
#endif

typedef __attribute__((ext_vector_type(16))) _Float16 v16h;
typedef __attribute__((ext_vector_type(8)))  _Float16 v8h;
typedef __attribute__((ext_vector_type(8)))  float    v8f;
typedef __attribute__((ext_vector_type(4)))  float    v4f;
typedef __attribute__((ext_vector_type(4)))  int      v4i;

#define T_STEPS 1000
#define BATCH   512
#define NIN     128
#define HDIM    250
#define HP      256      // padded hidden
#define NOUT    2
#define BETA    0.9f
#define THR     1.0f

#define BTILE   32                 // batch rows per workgroup
#define NWG     (BATCH / BTILE)    // 16 workgroups
#define NTHREADS 256               // 8 waves (wave32)
#define WAVES   8
#define MT      (BTILE / 16)       // 2 M-tiles
#define NTW     (HP / 16 / WAVES)  // 2 N-tiles per wave
#define K1      (NIN / 32)         // 4 k-steps (layer 1)
#define K2      (HP  / 32)         // 8 k-steps (layer 2)

// ---------------- gfx1250 async global->LDS staging ------------------------
#if __has_builtin(__builtin_amdgcn_global_load_async_to_lds_b128)
#define HAVE_ASYNC_LDS 1
#else
#define HAVE_ASYNC_LDS 0
#endif

typedef __attribute__((address_space(1))) v4i* as1_v4i;  // global 16B chunk
typedef __attribute__((address_space(3))) v4i* as3_v4i;  // LDS 16B chunk

// Copy 16 bytes global -> LDS.  Async variant is tracked by ASYNCcnt and
// bypasses VGPRs entirely (global_load_async_to_lds_b128).
__device__ __forceinline__ void copy16_g2l(const float* g, float* l) {
#if HAVE_ASYNC_LDS
  __builtin_amdgcn_global_load_async_to_lds_b128(
      (as1_v4i)(uintptr_t)g, (as3_v4i)(uintptr_t)l, 0, 0);
#else
  *(v4f*)l = *(const v4f*)g;
#endif
}

__device__ __forceinline__ void wait_async0() {
#if HAVE_ASYNC_LDS
#if __has_builtin(__builtin_amdgcn_s_wait_asynccnt)
  __builtin_amdgcn_s_wait_asynccnt(0);
#else
  asm volatile("s_wait_asynccnt 0x0" ::: "memory");
#endif
#endif
}

// ---------------- WMMA fragment helpers ------------------------------------

// 16x32 f16 A-fragment from row-major *f16* LDS (layer 2: spike buffer).
// Per ISA 7.12.2: per lane the fragment is two contiguous 8-half (16B)
// chunks 16 halves apart.
__device__ __forceinline__ v16h load_a_frag(const _Float16* p) {
  v8h c0 = *(const v8h*)(p);
  v8h c1 = *(const v8h*)(p + 16);
  return __builtin_shufflevector(c0, c1,
      0, 1, 2, 3, 4, 5, 6, 7, 8, 9, 10, 11, 12, 13, 14, 15);
}

// 16x32 f16 A-fragment from row-major *f32* LDS (layer 1: async-staged x),
// converting on the fly (v_cvt_pk pairs co-execute with WMMA).
__device__ __forceinline__ v16h load_a_frag_f32(const float* p) {
  v8f c0 = *(const v8f*)(p);
  v8f c1 = *(const v8f*)(p + 16);
  v16h a;
#pragma unroll
  for (int j = 0; j < 8; ++j) {
    a[j]     = (_Float16)c0[j];
    a[8 + j] = (_Float16)c1[j];
  }
  return a;
}

// 32x16 f16 B-fragment from row-major f32 W[k][n] (leading dim ldn).
// Lane = N column (lane&15); lane-half selects K base {0,8}; each lane
// holds K kbase+0..7 and kbase+16..23 of its column.
__device__ __forceinline__ v16h make_b_frag(const float* __restrict__ W,
                                            int ldn, int krows, int ncols,
                                            int n, int kbase) {
  v16h f;
#pragma unroll
  for (int j = 0; j < 16; ++j) {
    int k = kbase + (j < 8 ? j : j + 8);
    float v = (k < krows && n < ncols) ? W[k * ldn + n] : 0.0f;
    f[j] = (_Float16)v;
  }
  return f;
}

// LIF carry-in: C = beta*m + bias - (m > thr) * thr   (reset-by-subtract)
__device__ __forceinline__ v8f lif_carry(v8f m, float bias) {
  v8f c;
#pragma unroll
  for (int i = 0; i < 8; ++i) {
    float mm = m[i];
    c[i] = BETA * mm + bias - ((mm > THR) ? THR : 0.0f);
  }
  return c;
}

__global__ __launch_bounds__(NTHREADS, 1)
void snn_lif_persistent_gfx1250(const float* __restrict__ x,
                                const float* __restrict__ W1,
                                const float* __restrict__ b1,
                                const float* __restrict__ W2,
                                const float* __restrict__ b2,
                                const float* __restrict__ W3,
                                const float* __restrict__ b3,
                                float* __restrict__ out) {
  __shared__ float    xraw[BTILE * NIN]; // 16 KB: async-staged x_t tile (f32)
  __shared__ _Float16 s1[BTILE * HP];    // 16 KB: layer-1 spikes
  __shared__ _Float16 s2[BTILE * HP];    // 16 KB: layer-2 spikes
  __shared__ float    w3s[HDIM * NOUT];  //  2 KB: W3 cached

  const int tid  = threadIdx.x;
  const int lane = tid & 31;
  const int wave = tid >> 5;
  const int bg0  = blockIdx.x * BTILE;   // first batch row of this WG

  const int nlo     = lane & 15;               // N within tile / M within tile
  const int kb_half = (lane & 16) ? 8 : 0;     // B-frag K base per lane-half
  const int ka_half = (lane & 16) ? 8  : 0;    // A-frag K base per lane-half

  const float* x_wg = x + (size_t)bg0 * NIN;

  // ---- kick off async staging of x_0 (overlaps weight-fragment init) ------
#pragma unroll
  for (int it = 0; it < (BTILE * NIN / 4) / NTHREADS; ++it) {
    const int i = (tid + it * NTHREADS) * 4;
    copy16_g2l(x_wg + i, xraw + i);
  }

  // ---- one-time: weights -> VGPR B-fragments, biases -> per-lane scalars --
  v16h  w1f[NTW][K1];
  v16h  w2f[NTW][K2];
  float bias1[NTW], bias2[NTW];
#pragma unroll
  for (int ntl = 0; ntl < NTW; ++ntl) {
    const int ntg = wave * NTW + ntl;          // global N-tile (0..15)
    const int n   = ntg * 16 + nlo;            // global column
    bias1[ntl] = (n < HDIM) ? b1[n] : 0.0f;
    bias2[ntl] = (n < HDIM) ? b2[n] : 0.0f;
#pragma unroll
    for (int ks = 0; ks < K1; ++ks)
      w1f[ntl][ks] = make_b_frag(W1, HDIM, NIN, HDIM, n, ks * 32 + kb_half);
#pragma unroll
    for (int ks = 0; ks < K2; ++ks)
      w2f[ntl][ks] = make_b_frag(W2, HDIM, HDIM, HDIM, n, ks * 32 + kb_half);
  }
  for (int i = tid; i < HDIM * NOUT; i += NTHREADS) w3s[i] = W3[i];

  // ---- layer-3 state (one (batch,out) pair per thread for tid < 64) -------
  const int  b_l       = tid >> 1;
  const int  o_l       = tid & 1;
  const bool l3_active = (tid < BTILE * NOUT);
  float      bias3     = l3_active ? b3[o_l] : 0.0f;
  float      m3        = 0.0f;

  // ---- membrane state in accumulator registers ----------------------------
  v8f m1[MT][NTW] = {};
  v8f m2[MT][NTW] = {};

  wait_async0();
  __syncthreads();                                   // x_0 staged

#pragma unroll 1
  for (int t = 0; t < T_STEPS; ++t) {
    // ---- layer 1: [32,128] x [128,256], LIF, spikes -> s1 -----------------
#pragma unroll
    for (int mt = 0; mt < MT; ++mt) {
      const float* arow = xraw + (mt * 16 + nlo) * NIN + ka_half;
      v8f c[NTW];
#pragma unroll
      for (int ntl = 0; ntl < NTW; ++ntl)
        c[ntl] = lif_carry(m1[mt][ntl], bias1[ntl]);
      // double-buffered A: load k+1 while WMMAs consume k (kills WAR nops)
      v16h a = load_a_frag_f32(arow);
#pragma unroll
      for (int ks = 0; ks < K1; ++ks) {
        v16h an = a;
        if (ks + 1 < K1) an = load_a_frag_f32(arow + (ks + 1) * 32);
#pragma unroll
        for (int ntl = 0; ntl < NTW; ++ntl)
          c[ntl] = __builtin_amdgcn_wmma_f32_16x16x32_f16(
              false, a, false, w1f[ntl][ks], (short)0, c[ntl], false, false);
        a = an;
      }
      const int rbase = mt * 16 + ((lane >> 4) << 3);  // C/D layout rows
#pragma unroll
      for (int ntl = 0; ntl < NTW; ++ntl) {
        m1[mt][ntl] = c[ntl];
        const int col = (wave * NTW + ntl) * 16 + nlo;
#pragma unroll
        for (int i = 0; i < 8; ++i)
          s1[(rbase + i) * HP + col] =
              (c[ntl][i] > THR) ? (_Float16)1.0f : (_Float16)0.0f;
      }
    }
    __syncthreads();                                 // s1 ready; xraw(t) dead

    // ---- prefetch next x tile into LDS (overlaps layers 2 & 3) ------------
    if (t + 1 < T_STEPS) {
      const float* xn = x_wg + (size_t)(t + 1) * BATCH * NIN;
#pragma unroll
      for (int it = 0; it < (BTILE * NIN / 4) / NTHREADS; ++it) {
        const int i = (tid + it * NTHREADS) * 4;
        copy16_g2l(xn + i, xraw + i);
      }
    }

    // ---- layer 2: [32,256] x [256,256], LIF, spikes -> s2 -----------------
#pragma unroll
    for (int mt = 0; mt < MT; ++mt) {
      const _Float16* arow = s1 + (mt * 16 + nlo) * HP + ka_half;
      v8f c[NTW];
#pragma unroll
      for (int ntl = 0; ntl < NTW; ++ntl)
        c[ntl] = lif_carry(m2[mt][ntl], bias2[ntl]);
      v16h a = load_a_frag(arow);
#pragma unroll
      for (int ks = 0; ks < K2; ++ks) {
        v16h an = a;
        if (ks + 1 < K2) an = load_a_frag(arow + (ks + 1) * 32);
#pragma unroll
        for (int ntl = 0; ntl < NTW; ++ntl)
          c[ntl] = __builtin_amdgcn_wmma_f32_16x16x32_f16(
              false, a, false, w2f[ntl][ks], (short)0, c[ntl], false, false);
        a = an;
      }
      const int rbase = mt * 16 + ((lane >> 4) << 3);
#pragma unroll
      for (int ntl = 0; ntl < NTW; ++ntl) {
        m2[mt][ntl] = c[ntl];
        const int col = (wave * NTW + ntl) * 16 + nlo;
#pragma unroll
        for (int i = 0; i < 8; ++i)
          s2[(rbase + i) * HP + col] =
              (c[ntl][i] > THR) ? (_Float16)1.0f : (_Float16)0.0f;
      }
    }
    __syncthreads();                                 // s2 ready

    // ---- layer 3: [32,256] x [250,2] via VALU dots, LIF, write outputs ----
    if (l3_active) {
      float cur = bias3;
      const _Float16* srow = s2 + b_l * HP;
#pragma unroll 2
      for (int k = 0; k < HDIM; ++k)
        cur += (float)srow[k] * w3s[k * NOUT + o_l];
      const float reset = (m3 > THR) ? THR : 0.0f;
      m3 = BETA * m3 + cur - reset;
      const float s3 = (m3 > THR) ? 1.0f : 0.0f;
      const size_t idx = (size_t)t * BATCH * NOUT + (size_t)(bg0 + b_l) * NOUT + o_l;
      out[idx]                                  = s3;  // spikes
      out[(size_t)T_STEPS * BATCH * NOUT + idx] = m3;  // mem_rec
    }
    wait_async0();                                   // x_{t+1} staged
    __syncthreads();
  }
}

extern "C" void kernel_launch(void* const* d_in, const int* in_sizes, int n_in,
                              void* d_out, int out_size, void* d_ws, size_t ws_size,
                              hipStream_t stream) {
  (void)in_sizes; (void)n_in; (void)out_size; (void)d_ws; (void)ws_size;
  const float* x  = (const float*)d_in[0];
  const float* W1 = (const float*)d_in[1];
  const float* b1 = (const float*)d_in[2];
  const float* W2 = (const float*)d_in[3];
  const float* b2 = (const float*)d_in[4];
  const float* W3 = (const float*)d_in[5];
  const float* b3 = (const float*)d_in[6];
  float* out = (float*)d_out;
  snn_lif_persistent_gfx1250<<<NWG, NTHREADS, 0, stream>>>(
      x, W1, b1, W2, b2, W3, b3, out);
}
